// LSTM_Mod2_11398843203682
// MI455X (gfx1250) — matrix operand order repaired and checked by reading, not compiled, not run on hardware
//
#include <hip/hip_runtime.h>
#include <hip/hip_bf16.h>
#include <stdint.h>

#define T_STEPS 1024
#define BSZ 64
#define HDIM 1024
#define VDIM 128

#define NGATE 64          // gate workgroups, each owns 16 hidden columns
#define NPROJ 8           // projection workgroups, each owns 16 vocab columns
#define NWG   (NGATE + NPROJ)
#define JW    16          // hidden columns per gate WG
#define PITCH (HDIM + 8)  // bf16 elements per LDS row (1032 -> 2064 B, spreads banks)

// ---- workspace byte layout ----
#define WHH_OFF   0ull                                            // 4H*H bf16 = 8 MB
#define WLIN_OFF  (WHH_OFF  + 4ull*HDIM*HDIM*2ull)                // V*H bf16  = 256 KB
#define BIAS_OFF  (WLIN_OFF + (unsigned long long)VDIM*HDIM*2ull) // 4H f32    = 16 KB
#define HBUF_OFF  (BIAS_OFF + 4ull*HDIM*4ull)                     // 2*BS*H bf16 = 256 KB
#define CNT_OFF   (HBUF_OFF + 2ull*BSZ*HDIM*2ull)                 // barrier counter

typedef __attribute__((ext_vector_type(16))) __bf16 v16bf;
typedef __attribute__((ext_vector_type(8)))  float  v8f;

struct alignas(16) B128 { unsigned int d[4]; };
struct B256 { B128 lo, hi; };

static __device__ __forceinline__ unsigned short f2bf(float f) {
  unsigned u = __builtin_bit_cast(unsigned, f);
  unsigned r = ((u >> 16) & 1u) + 0x7fffu;     // round-to-nearest-even
  return (unsigned short)((u + r) >> 16);
}
static __device__ __forceinline__ float sigm(float x)  { return 1.f / (1.f + __expf(-x)); }
static __device__ __forceinline__ float tanhf_(float x){ return 2.f / (1.f + __expf(-2.f*x)) - 1.f; }

static __device__ __forceinline__ v16bf load_frag_g(const unsigned short* p0,
                                                    const unsigned short* p1) {
  B256 t;
  t.lo = *(const B128*)p0;
  t.hi = *(const B128*)p1;
  return __builtin_bit_cast(v16bf, t);
}

// ---------------- prep: bf16 weight copies, fused bias, zero h, reset barrier ----
__global__ void lstm_prep(const float* __restrict__ Whh, const float* __restrict__ Wlin,
                          const float* __restrict__ bih, const float* __restrict__ bhh,
                          unsigned char* __restrict__ ws) {
  unsigned short* whh  = (unsigned short*)(ws + WHH_OFF);
  unsigned short* wlin = (unsigned short*)(ws + WLIN_OFF);
  float*          bias = (float*)(ws + BIAS_OFF);
  unsigned short* hbuf = (unsigned short*)(ws + HBUF_OFF);
  unsigned*       cnt  = (unsigned*)(ws + CNT_OFF);
  int idx = blockIdx.x * blockDim.x + threadIdx.x;
  int str = gridDim.x * blockDim.x;
  for (int i = idx; i < 4*HDIM*HDIM; i += str) whh[i]  = f2bf(Whh[i]);
  for (int i = idx; i < VDIM*HDIM;  i += str) wlin[i] = f2bf(Wlin[i]);
  for (int i = idx; i < 4*HDIM;     i += str) bias[i] = bih[i] + bhh[i];
  for (int i = idx; i < 2*BSZ*HDIM; i += str) hbuf[i] = 0;
  if (idx == 0) *cnt = 0u;
}

// ---------------- persistent LSTM scan ----------------
__global__ void __launch_bounds__(128, 1)
lstm_scan(const float* __restrict__ sentence, const float* __restrict__ Wih,
          const float* __restrict__ blin, float* __restrict__ out,
          unsigned char* __restrict__ ws) {
  extern __shared__ unsigned char smem[];   // gate WGs: 64 x PITCH bf16 W_hh slice

  const unsigned short* whh  = (const unsigned short*)(ws + WHH_OFF);
  const unsigned short* wlin = (const unsigned short*)(ws + WLIN_OFF);
  const float*          bias = (const float*)(ws + BIAS_OFF);
  unsigned short*       hbuf = (unsigned short*)(ws + HBUF_OFF);
  unsigned*             cnt  = (unsigned*)(ws + CNT_OFF);

  const int wg     = blockIdx.x;
  const bool isGate = (wg < NGATE);
  const int tid  = threadIdx.x;
  const int lane = tid & 31;
  const int wave = tid >> 5;          // 4 waves per WG
  const int hi   = lane >> 4;         // half-wave select
  const int ln   = lane & 15;
  const int b0   = wave * 16;         // this wave's batch-row tile

  // ---- one-time LDS fill: this WG's 64 gate rows of W_hh (bf16) ----
  if (isGate) {
    for (int chunk = tid; chunk < 64 * 128; chunk += 128) {
      int row = chunk >> 7;           // 0..63 : nt*16 + nl
      int cc  = chunk & 127;          // 16-byte chunk within row
      int nt  = row >> 4, nl = row & 15;
      const B128* src = (const B128*)(whh + (size_t)(nt*HDIM + wg*JW + nl) * HDIM) + cc;
      *(B128*)(smem + (size_t)row * (PITCH*2) + (size_t)cc * 16) = *src;
    }
  }
  __syncthreads();

  // ---- per-lane constants (column index n = ln is fixed per lane) ----
  const int jglob = wg * JW + ln;                 // hidden column (gate WGs)
  float wih_r[4], bias_r[4];
  if (isGate) {
    #pragma unroll
    for (int nt = 0; nt < 4; ++nt) {
      wih_r[nt]  = Wih [nt*HDIM + jglob];
      bias_r[nt] = bias[nt*HDIM + jglob];
    }
  }
  const int pv = (wg - NGATE) * 16 + ln;          // vocab column (proj WGs)
  float blin_r = isGate ? 0.f : blin[pv];

  float c_reg[8];                                  // cell state lives in VGPRs
  #pragma unroll
  for (int r = 0; r < 8; ++r) c_reg[r] = 0.f;

  for (int t = 0; t <= T_STEPS; ++t) {
    const unsigned short* hrd = hbuf + (size_t)(t & 1) * BSZ * HDIM;

    if (isGate && t < T_STEPS) {
      unsigned short* hwr = hbuf + (size_t)((t + 1) & 1) * BSZ * HDIM;
      v8f acc[4];
      #pragma unroll
      for (int nt = 0; nt < 4; ++nt) acc[nt] = (v8f){0.f,0.f,0.f,0.f,0.f,0.f,0.f,0.f};

      #pragma unroll 2
      for (int k0 = 0; k0 < HDIM; k0 += 32) {
        // A: h fragment 16x32 bf16 (rows b0..b0+15)
        const unsigned short* ap = hrd + (size_t)(b0 + ln) * HDIM + k0 + hi * 8;
        v16bf a = load_frag_g(ap, ap + 16);
        #pragma unroll
        for (int nt = 0; nt < 4; ++nt) {            // nt = gate i,f,g,o
          const unsigned char* bp = smem + (size_t)(nt*16 + ln) * (PITCH*2)
                                         + (size_t)(k0 + hi*8) * 2;
          B256 tb; tb.lo = *(const B128*)bp; tb.hi = *(const B128*)(bp + 32);
          v16bf b = __builtin_bit_cast(v16bf, tb);
          acc[nt] = __builtin_amdgcn_wmma_f32_16x16x32_bf16(
              false, a, false, b, (short)0, acc[nt], false, false);
        }
      }

      // scalar-input term + fused bias, then gate nonlinearities; c stays in regs
      #pragma unroll
      for (int r = 0; r < 8; ++r) {
        const int b = b0 + r + 8 * hi;
        const float xv = sentence[t * BSZ + b];
        const float gi = sigm (acc[0][r] + xv * wih_r[0] + bias_r[0]);
        const float gf = sigm (acc[1][r] + xv * wih_r[1] + bias_r[1]);
        const float gg = tanhf_(acc[2][r] + xv * wih_r[2] + bias_r[2]);
        const float go = sigm (acc[3][r] + xv * wih_r[3] + bias_r[3]);
        const float c  = gf * c_reg[r] + gi * gg;
        c_reg[r] = c;
        hwr[(size_t)b * HDIM + jglob] = f2bf(go * tanhf_(c));
      }
    }

    if (!isGate && t > 0) {                        // out_{t-1} = h_{t-1} @ W_lin.T + b
      v8f pacc = (v8f){0.f,0.f,0.f,0.f,0.f,0.f,0.f,0.f};
      #pragma unroll 2
      for (int k0 = 0; k0 < HDIM; k0 += 32) {
        const unsigned short* ap = hrd + (size_t)(b0 + ln) * HDIM + k0 + hi * 8;
        v16bf a = load_frag_g(ap, ap + 16);
        const unsigned short* bp = wlin + (size_t)pv * HDIM + k0 + hi * 8;
        v16bf b = load_frag_g(bp, bp + 16);
        pacc = __builtin_amdgcn_wmma_f32_16x16x32_bf16(
            false, a, false, b, (short)0, pacc, false, false);
      }
      #pragma unroll
      for (int r = 0; r < 8; ++r) {
        const int b = b0 + r + 8 * hi;
        out[((size_t)(t - 1) * BSZ + b) * VDIM + pv] = pacc[r] + blin_r;
      }
    }

    if (t < T_STEPS) {                             // grid barrier between steps
      __builtin_amdgcn_fence(__ATOMIC_RELEASE, "agent");
      __syncthreads();
      if (tid == 0) {
        __hip_atomic_fetch_add(cnt, 1u, __ATOMIC_RELAXED, __HIP_MEMORY_SCOPE_AGENT);
        const unsigned target = (unsigned)(t + 1) * NWG;
        while (__hip_atomic_load(cnt, __ATOMIC_RELAXED, __HIP_MEMORY_SCOPE_AGENT) < target)
          __builtin_amdgcn_s_sleep(1);
      }
      __syncthreads();
      __builtin_amdgcn_fence(__ATOMIC_ACQUIRE, "agent");
    }
  }
}

extern "C" void kernel_launch(void* const* d_in, const int* in_sizes, int n_in,
                              void* d_out, int out_size, void* d_ws, size_t ws_size,
                              hipStream_t stream) {
  const float* sentence = (const float*)d_in[0];   // [T, BS]
  const float* Wih      = (const float*)d_in[1];   // [4H, 1]
  const float* Whh      = (const float*)d_in[2];   // [4H, H]
  const float* bih      = (const float*)d_in[3];   // [4H]
  const float* bhh      = (const float*)d_in[4];   // [4H]
  const float* Wlin     = (const float*)d_in[5];   // [V, H]
  const float* blin     = (const float*)d_in[6];   // [V]
  float* out            = (float*)d_out;           // [T, BS, V]
  unsigned char* ws     = (unsigned char*)d_ws;

  lstm_prep<<<2048, 256, 0, stream>>>(Whh, Wlin, bih, bhh, ws);
  const size_t smem = (size_t)64 * (PITCH * 2);    // 132,096 B W_hh slice in LDS
  lstm_scan<<<NWG, 128, smem, stream>>>(sentence, Wih, blin, out, ws);
}
